// SubBlock_49340584297220
// MI455X (gfx1250) — matrix operand order repaired and checked by reading, not compiled, not run on hardware
//
#include <hip/hip_runtime.h>
#include <math.h>

// ---------------------------------------------------------------------------
// MI455X / gfx1250 implementation of the AVD equivariant block.
// All channel contractions (linear mixes, low-rank tensor prods, K=7 convs)
// are routed through V_WMMA_F32_16X16X32_BF16: bf16 operands, f32 accumulate.
// ~120 GFLOP vs ~400MB HBM traffic -> matrix-core bound, so weights are
// pre-transposed/converted to bf16 once (4.0MB) and each wave computes a
// 16x32 tile (one A fragment feeds two back-to-back WMMAs).
// ---------------------------------------------------------------------------

typedef __bf16 bf16_t;
typedef __attribute__((ext_vector_type(16))) __bf16 v16bf;
typedef __attribute__((ext_vector_type(8)))  float  v8f;

#define BB   8
#define LL   2048
#define NPOS (BB * LL)      // 16384 positions
#define DA   256
#define DV   128
#define DDIM 64
#define RR   32
#define KW   7

// gemm flags
#define GF_ACC   1   // D += A@B
#define GF_CIN   2   // D  = Cin + A@B
// conv flags
#define CF_BIAS  1
#define CF_LRELU 2
#define CF_RES   4

__device__ __forceinline__ float lrelu_f(float x) { return x > 0.f ? x : 0.1f * x; }

__device__ __forceinline__ v16bf pack_a(float4 a0, float4 a1, float4 a2, float4 a3)
{
    v16bf af;
    af[0]  = (bf16_t)a0.x; af[1]  = (bf16_t)a0.y; af[2]  = (bf16_t)a0.z; af[3]  = (bf16_t)a0.w;
    af[4]  = (bf16_t)a1.x; af[5]  = (bf16_t)a1.y; af[6]  = (bf16_t)a1.z; af[7]  = (bf16_t)a1.w;
    af[8]  = (bf16_t)a2.x; af[9]  = (bf16_t)a2.y; af[10] = (bf16_t)a2.z; af[11] = (bf16_t)a2.w;
    af[12] = (bf16_t)a3.x; af[13] = (bf16_t)a3.y; af[14] = (bf16_t)a3.z; af[15] = (bf16_t)a3.w;
    return af;
}

// ---------------------------------------------------------------------------
// Weight prep: W[(ntap), Kc, Nc] f32 row-major -> Wt[(ntap), Nc, Kc] bf16.
// In the transposed layout a B fragment (lane=col, 16 consecutive k) is one
// contiguous 32-byte load.
// ---------------------------------------------------------------------------
__global__ void prep_wt_kernel(const float* __restrict__ W, bf16_t* __restrict__ Wt,
                               int Kc, int Nc, int ntap)
{
    int idx = blockIdx.x * blockDim.x + threadIdx.x;
    int per = Kc * Nc;
    if (idx >= ntap * per) return;
    int tap = idx / per, r = idx % per;
    int n = r / Kc, k = r % Kc;
    Wt[idx] = (bf16_t)W[(size_t)tap * per + (size_t)k * Nc + n];
}

// ---------------------------------------------------------------------------
// GEMM:  D[M,Nc] = A[M,Kc] @ Bt^T (+Cin) (+D).  A f32 row-major, Bt bf16
// (Nc x Kc).  128 threads = 4 waves, each wave one 16(M) x 32(N) tile.
// Fragment layouts used here (A and B share one k order):
//   A (16x32): lane m + 16*h, elem e -> k = 16h + e   (contiguous in A row)
//   B (32x16): lane n + 16*h, elem e -> k = 16h + e   (contiguous in Bt)
//   C/D:       VGPR j -> M = j + 8h, N = lane&15
// ---------------------------------------------------------------------------
template<int FLAGS>
__global__ __launch_bounds__(128)
void gemm_wmma(const float* __restrict__ A, const bf16_t* __restrict__ Bt,
               const float* __restrict__ Cin, float* __restrict__ D,
               int Kc, int Nc)
{
    const int lane = threadIdx.x & 31;
    const int wv   = threadIdx.x >> 5;
    const int h    = lane >> 4;
    const int mn   = lane & 15;
    const int row0 = (blockIdx.x * 4 + wv) * 16;
    const int col0 = blockIdx.y * 32;

    const float*  Arow = A  + (size_t)(row0 + mn) * Kc + 16 * h;
    const bf16_t* B0   = Bt + (size_t)(col0 + mn) * Kc + 16 * h;
    const bf16_t* B1   = B0 + (size_t)16 * Kc;

    v8f acc0 = {}, acc1 = {};
    for (int kk = 0; kk < Kc; kk += 32) {
        __builtin_prefetch(Arow + kk + 32, 0, 0);   // global_prefetch_b8
        float4 a0 = *(const float4*)(Arow + kk);
        float4 a1 = *(const float4*)(Arow + kk + 4);
        float4 a2 = *(const float4*)(Arow + kk + 8);
        float4 a3 = *(const float4*)(Arow + kk + 12);
        v16bf af = pack_a(a0, a1, a2, a3);
        v16bf b0 = *(const v16bf*)(B0 + kk);
        v16bf b1 = *(const v16bf*)(B1 + kk);
        acc0 = __builtin_amdgcn_wmma_f32_16x16x32_bf16(false, af, false, b0,
                                                       (short)0, acc0, false, false);
        acc1 = __builtin_amdgcn_wmma_f32_16x16x32_bf16(false, af, false, b1,
                                                       (short)0, acc1, false, false);
    }
#pragma unroll
    for (int j = 0; j < 8; ++j) {
        int    r = row0 + j + 8 * h;
        size_t o = (size_t)r * Nc + col0 + mn;
        float v0 = acc0[j], v1 = acc1[j];
        if constexpr (FLAGS & GF_CIN) { v0 += Cin[o]; v1 += Cin[o + 16]; }
        if constexpr (FLAGS & GF_ACC) { v0 += D[o];   v1 += D[o + 16];   }
        D[o] = v0; D[o + 16] = v1;
    }
}

// ---------------------------------------------------------------------------
// Conv1d (SAME, K=7) over L of planar X[S][L][C] as 7 shifted GEMM
// accumulations; Wt prepped bf16 (tap, Cout, Cin). 4 waves/WG, each wave
// one 16(L) x 32(C) tile. Fused bias / lrelu / residual at compile time.
// ---------------------------------------------------------------------------
template<int FLAGS>
__global__ __launch_bounds__(128)
void conv_wmma(const float* __restrict__ X, const bf16_t* __restrict__ Wt,
               const float* __restrict__ bias, const float* __restrict__ Res,
               float* __restrict__ Y, int C)
{
    const int lane = threadIdx.x & 31;
    const int wv   = threadIdx.x >> 5;
    const int h    = lane >> 4;
    const int mn   = lane & 15;
    const int tilesPerSeq = LL / 64;
    const int s    = blockIdx.x / tilesPerSeq;
    const int l0   = (blockIdx.x % tilesPerSeq) * 64 + wv * 16;
    const int col0 = blockIdx.y * 32;

    const int    lrow   = l0 + mn;
    const size_t seqOff = (size_t)s * LL * C;

    v8f acc0 = {}, acc1 = {};
    for (int tap = 0; tap < KW; ++tap) {
        int  p     = lrow + tap - 3;
        bool valid = (p >= 0) && (p < LL);
        const float*  Arow = X + seqOff + (size_t)(valid ? p : 0) * C + 16 * h;
        const bf16_t* B0   = Wt + (size_t)tap * C * C + (size_t)(col0 + mn) * C + 16 * h;
        const bf16_t* B1   = B0 + (size_t)16 * C;
        for (int kk = 0; kk < C; kk += 32) {
            float4 a0, a1, a2, a3;
            if (valid) {
                a0 = *(const float4*)(Arow + kk);
                a1 = *(const float4*)(Arow + kk + 4);
                a2 = *(const float4*)(Arow + kk + 8);
                a3 = *(const float4*)(Arow + kk + 12);
            } else {
                a0 = a1 = a2 = a3 = make_float4(0.f, 0.f, 0.f, 0.f);
            }
            v16bf af = pack_a(a0, a1, a2, a3);
            v16bf b0 = *(const v16bf*)(B0 + kk);
            v16bf b1 = *(const v16bf*)(B1 + kk);
            acc0 = __builtin_amdgcn_wmma_f32_16x16x32_bf16(false, af, false, b0,
                                                           (short)0, acc0, false, false);
            acc1 = __builtin_amdgcn_wmma_f32_16x16x32_bf16(false, af, false, b1,
                                                           (short)0, acc1, false, false);
        }
    }
    float bv0 = 0.f, bv1 = 0.f;
    if constexpr (FLAGS & CF_BIAS) { bv0 = bias[col0 + mn]; bv1 = bias[col0 + 16 + mn]; }
#pragma unroll
    for (int j = 0; j < 8; ++j) {
        int    r = l0 + j + 8 * h;
        size_t o = seqOff + (size_t)r * C + col0 + mn;
        float v0 = acc0[j] + bv0, v1 = acc1[j] + bv1;
        if constexpr (FLAGS & CF_LRELU) { v0 = lrelu_f(v0); v1 = lrelu_f(v1); }
        if constexpr (FLAGS & CF_RES)   { v0 += Res[o];     v1 += Res[o + 16]; }
        Y[o] = v0; Y[o + 16] = v1;
    }
}

// ---------------------------------------------------------------------------
// Elementwise / pack / norm kernels
// ---------------------------------------------------------------------------
__global__ void pack_v_kernel(const float* __restrict__ xv, float* __restrict__ vP)
{
    int idx = blockIdx.x * blockDim.x + threadIdx.x;       // n*DV + c
    if (idx >= NPOS * DV) return;
#pragma unroll
    for (int i = 0; i < 3; ++i)
        vP[(size_t)i * NPOS * DV + idx] = xv[(size_t)idx * 3 + i];
}

__global__ void pack_d_kernel(const float* __restrict__ xd, float* __restrict__ dP,
                              float* __restrict__ dPT, float* __restrict__ tr)
{
    int idx = blockIdx.x * blockDim.x + threadIdx.x;       // n*DD + c
    if (idx >= NPOS * DDIM) return;
    const float* src = xd + (size_t)idx * 9;
    float t = 0.f;
#pragma unroll
    for (int i = 0; i < 3; ++i)
#pragma unroll
        for (int j = 0; j < 3; ++j) {
            float v = src[i * 3 + j];
            dP [(size_t)(i * 3 + j) * NPOS * DDIM + idx] = v;
            dPT[(size_t)(j * 3 + i) * NPOS * DDIM + idx] = v;
            if (i == j) t += v;
        }
    tr[idx] = t;
}

__global__ void add_diag_kernel(float* __restrict__ dL, const float* __restrict__ tad)
{
    int idx = blockIdx.x * blockDim.x + threadIdx.x;       // n*DD + c
    if (idx >= NPOS * DDIM) return;
    float t = tad[idx];
    dL[(size_t)0 * NPOS * DDIM + idx] += t;
    dL[(size_t)4 * NPOS * DDIM + idx] += t;
    dL[(size_t)8 * NPOS * DDIM + idx] += t;
}

__global__ void bilin_a_kernel(const float* __restrict__ pa1, const float* __restrict__ pa2,
                               const float* __restrict__ pv1, const float* __restrict__ pv2,
                               const float* __restrict__ pd1, const float* __restrict__ pd2,
                               float* __restrict__ sa, float* __restrict__ sv,
                               float* __restrict__ sd)
{
    int idx = blockIdx.x * blockDim.x + threadIdx.x;       // n*RR + r
    if (idx >= NPOS * RR) return;
    const size_t NR = (size_t)NPOS * RR;
    sa[idx] = pa1[idx] * pa2[idx];
    float v = 0.f;
#pragma unroll
    for (int i = 0; i < 3; ++i) v += pv1[i * NR + idx] * pv2[i * NR + idx];
    sv[idx] = v;
    float d = 0.f;
#pragma unroll
    for (int q = 0; q < 9; ++q) d += pd1[q * NR + idx] * pd2[q * NR + idx];
    sd[idx] = d;
}

__global__ void bilin_v_kernel(const float* __restrict__ pa1, const float* __restrict__ pv2,
                               const float* __restrict__ pd1,
                               float* __restrict__ tva, float* __restrict__ tvd)
{
    int idx = blockIdx.x * blockDim.x + threadIdx.x;
    if (idx >= NPOS * RR) return;
    const size_t NR = (size_t)NPOS * RR;
    float a = pa1[idx];
#pragma unroll
    for (int i = 0; i < 3; ++i) {
        tva[i * NR + idx] = a * pv2[i * NR + idx];
        float s = 0.f;
#pragma unroll
        for (int j = 0; j < 3; ++j)
            s += pd1[(size_t)(i * 3 + j) * NR + idx] * pv2[j * NR + idx];
        tvd[i * NR + idx] = s;
    }
}

__global__ void bilin_d_kernel(const float* __restrict__ pa1,
                               const float* __restrict__ pv1, const float* __restrict__ pv2,
                               const float* __restrict__ pd1, const float* __restrict__ pd2,
                               float* __restrict__ tda, float* __restrict__ tdv,
                               float* __restrict__ tdd)
{
    int idx = blockIdx.x * blockDim.x + threadIdx.x;
    if (idx >= NPOS * RR) return;
    const size_t NR = (size_t)NPOS * RR;
    float a = pa1[idx];
#pragma unroll
    for (int i = 0; i < 3; ++i)
#pragma unroll
        for (int j = 0; j < 3; ++j) {
            int q = i * 3 + j;
            tda[(size_t)q * NR + idx] = a * pd2[(size_t)q * NR + idx];
            tdv[(size_t)q * NR + idx] = pv1[i * NR + idx] * pv2[j * NR + idx];
            float s = 0.f;
#pragma unroll
            for (int k = 0; k < 3; ++k)
                s += pd1[(size_t)(i * 3 + k) * NR + idx] * pd2[(size_t)(k * 3 + j) * NR + idx];
            tdd[(size_t)q * NR + idx] = s;
        }
}

// equivariant sigmoid gate, in place on planar X[(ncomp, NPOS, C)]
__global__ void tsig_kernel(float* __restrict__ X, int C, int ncomp)
{
    int idx = blockIdx.x * blockDim.x + threadIdx.x;       // n*C + c
    if (idx >= NPOS * C) return;
    const size_t stride = (size_t)NPOS * C;
    float s = 1e-6f;
    for (int i = 0; i < ncomp; ++i) {
        float v = X[i * stride + idx];
        s += v * v;
    }
    float nrm = sqrtf(s);
    float g   = 1.f / (1.f + __expf(-nrm));
    for (int i = 0; i < ncomp; ++i)
        X[i * stride + idx] *= g;
}

__global__ void gn_a_kernel(const float* __restrict__ X, const float* __restrict__ res,
                            const float* __restrict__ gamma, const float* __restrict__ beta,
                            float* __restrict__ out)
{
    int idx = blockIdx.x * blockDim.x + threadIdx.x;       // n*8 + g
    if (idx >= NPOS * 8) return;
    int n = idx >> 3, g = idx & 7;
    const int GC = DA / 8;                                 // 32 channels per group
    const float* xp = X + (size_t)n * DA + g * GC;
    float mean = 0.f;
    for (int c = 0; c < GC; ++c) mean += xp[c];
    mean *= (1.f / GC);
    float ms = 0.f;
    for (int c = 0; c < GC; ++c) { float d = xp[c] - mean; ms += d * d; }
    float inv = rsqrtf(ms * (1.f / GC) + 1e-5f);
    const float* rp = res + (size_t)n * DA + g * GC;
    float*       op = out + (size_t)n * DA + g * GC;
    for (int c = 0; c < GC; ++c) {
        int ch = g * GC + c;
        op[c] = rp[c] + (xp[c] - mean) * inv * gamma[ch] + beta[ch];
    }
}

__global__ void gn_v_kernel(const float* __restrict__ X, const float* __restrict__ res,
                            const float* __restrict__ gamma, float* __restrict__ out)
{
    int idx = blockIdx.x * blockDim.x + threadIdx.x;       // n*8 + g
    if (idx >= NPOS * 8) return;
    int n = idx >> 3, g = idx & 7;
    const int GC = DV / 8;                                 // 16
    const size_t stride = (size_t)NPOS * DV;
    float ms = 0.f;
    for (int c = 0; c < GC; ++c) {
        int ch = g * GC + c;
        for (int i = 0; i < 3; ++i) {
            float v = X[i * stride + (size_t)n * DV + ch];
            ms += v * v;
        }
    }
    float inv = rsqrtf(ms * (1.f / (GC * 3)) + 1e-5f);
    for (int c = 0; c < GC; ++c) {
        int ch = g * GC + c;
        for (int i = 0; i < 3; ++i) {
            float  y = X[i * stride + (size_t)n * DV + ch] * inv * gamma[ch];
            size_t o = (size_t)n * DV * 3 + ch * 3 + i;
            out[o] = res[o] + y;
        }
    }
}

__global__ void gn_d_kernel(const float* __restrict__ X, const float* __restrict__ res,
                            const float* __restrict__ gamma, float* __restrict__ out)
{
    int idx = blockIdx.x * blockDim.x + threadIdx.x;       // n*8 + g
    if (idx >= NPOS * 8) return;
    int n = idx >> 3, g = idx & 7;
    const int GC = DDIM / 8;                               // 8
    const size_t stride = (size_t)NPOS * DDIM;
    float ms = 0.f;
    for (int c = 0; c < GC; ++c) {
        int ch = g * GC + c;
        for (int q = 0; q < 9; ++q) {
            float v = X[q * stride + (size_t)n * DDIM + ch];
            ms += v * v;
        }
    }
    float inv = rsqrtf(ms * (1.f / (GC * 9)) + 1e-5f);
    for (int c = 0; c < GC; ++c) {
        int ch = g * GC + c;
        for (int q = 0; q < 9; ++q) {
            float  y = X[q * stride + (size_t)n * DDIM + ch] * inv * gamma[ch];
            size_t o = (size_t)n * DDIM * 9 + ch * 9 + q;
            out[o] = res[o] + y;
        }
    }
}

// ---------------------------------------------------------------------------
// Host-side orchestration
// ---------------------------------------------------------------------------
extern "C" void kernel_launch(void* const* d_in, const int* in_sizes, int n_in,
                              void* d_out, int out_size, void* d_ws, size_t ws_size,
                              hipStream_t stream)
{
    (void)in_sizes; (void)n_in; (void)out_size;
    const float* x_a  = (const float*)d_in[0];
    const float* x_v  = (const float*)d_in[1];
    const float* x_d  = (const float*)d_in[2];
    const float* Waa  = (const float*)d_in[3];
    const float* Wda  = (const float*)d_in[4];
    const float* Wvv  = (const float*)d_in[5];
    const float* Wdd  = (const float*)d_in[6];
    const float* Wdt  = (const float*)d_in[7];
    const float* Wad  = (const float*)d_in[8];
    const float* A1   = (const float*)d_in[9];
    const float* A2   = (const float*)d_in[10];
    const float* V1   = (const float*)d_in[11];
    const float* V2   = (const float*)d_in[12];
    const float* D1   = (const float*)d_in[13];
    const float* D2   = (const float*)d_in[14];
    const float* Oaaa = (const float*)d_in[15];
    const float* Oavv = (const float*)d_in[16];
    const float* Oadd = (const float*)d_in[17];
    const float* Ovav = (const float*)d_in[18];
    const float* Ovdv = (const float*)d_in[19];
    const float* Odad = (const float*)d_in[20];
    const float* Odvv = (const float*)d_in[21];
    const float* Oddd = (const float*)d_in[22];
    const float* Wa1  = (const float*)d_in[23];
    const float* ba1  = (const float*)d_in[24];
    const float* Wa2  = (const float*)d_in[25];
    const float* ba2  = (const float*)d_in[26];
    const float* Wa3  = (const float*)d_in[27];
    const float* ba3  = (const float*)d_in[28];
    const float* Wv1  = (const float*)d_in[29];
    const float* Wv2  = (const float*)d_in[30];
    const float* Wv3  = (const float*)d_in[31];
    const float* Wd1  = (const float*)d_in[32];
    const float* Wd2  = (const float*)d_in[33];
    const float* Wd3  = (const float*)d_in[34];
    const float* ga   = (const float*)d_in[35];
    const float* bA   = (const float*)d_in[36];
    const float* gv   = (const float*)d_in[37];
    const float* gd   = (const float*)d_in[38];

    // ---- workspace layout: f32 activations, then bf16 prepped weights ----
    float* ws  = (float*)d_ws;
    size_t off = 0;
    const size_t NP = (size_t)NPOS;
    auto alloc = [&](size_t nf) { float* p = ws + off; off += nf; return p; };
    float* tr  = alloc(NP * DDIM);
    float* aL  = alloc(NP * DA);
    float* vP  = alloc(NP * DV * 3);
    float* vL  = alloc(NP * DV * 3);
    float* dP  = alloc(NP * DDIM * 9);
    float* dPT = alloc(NP * DDIM * 9);
    float* dL  = alloc(NP * DDIM * 9);
    float* tad = alloc(NP * DDIM);
    float* pa1 = alloc(NP * RR);     float* pa2 = alloc(NP * RR);
    float* pv1 = alloc(NP * RR * 3); float* pv2 = alloc(NP * RR * 3);
    float* pd1 = alloc(NP * RR * 9); float* pd2 = alloc(NP * RR * 9);
    float* sa  = alloc(NP * RR); float* sv = alloc(NP * RR); float* sd = alloc(NP * RR);
    float* tva = alloc(NP * RR * 3); float* tvd = alloc(NP * RR * 3);
    float* tda = alloc(NP * RR * 9); float* tdv = alloc(NP * RR * 9); float* tdd = alloc(NP * RR * 9);
    float* v1  = alloc(NP * DV * 3);
    float* d1  = alloc(NP * DDIM * 9);
    float* a1  = vP;   // vP dead after the Wvv GEMM
    float* t1  = dP;   // dP/dPT dead after the linear mix -> conv scratch
    float* t2  = dPT;

    bf16_t* wb   = (bf16_t*)(ws + off);
    size_t  boff = 0;
    auto balloc = [&](size_t ne) { bf16_t* p = wb + boff; boff += (ne + 63) & ~(size_t)63; return p; };
    bf16_t* Waa_t  = balloc(DA * DA);
    bf16_t* Wda_t  = balloc(DDIM * DA);
    bf16_t* Wvv_t  = balloc(DV * DV);
    bf16_t* Wdd_t  = balloc(DDIM * DDIM);
    bf16_t* Wdt_t  = balloc(DDIM * DDIM);
    bf16_t* Wad_t  = balloc(DA * DDIM);
    bf16_t* A1t    = balloc(DA * RR);   bf16_t* A2t = balloc(DA * RR);
    bf16_t* V1t    = balloc(DV * RR);   bf16_t* V2t = balloc(DV * RR);
    bf16_t* D1t    = balloc(DDIM * RR); bf16_t* D2t = balloc(DDIM * RR);
    bf16_t* Oaaa_t = balloc(RR * DA);   bf16_t* Oavv_t = balloc(RR * DA);  bf16_t* Oadd_t = balloc(RR * DA);
    bf16_t* Ovav_t = balloc(RR * DV);   bf16_t* Ovdv_t = balloc(RR * DV);
    bf16_t* Odad_t = balloc(RR * DDIM); bf16_t* Odvv_t = balloc(RR * DDIM); bf16_t* Oddd_t = balloc(RR * DDIM);
    bf16_t* Wa1t = balloc(KW * DA * DA);     bf16_t* Wa2t = balloc(KW * DA * DA);     bf16_t* Wa3t = balloc(KW * DA * DA);
    bf16_t* Wv1t = balloc(KW * DV * DV);     bf16_t* Wv2t = balloc(KW * DV * DV);     bf16_t* Wv3t = balloc(KW * DV * DV);
    bf16_t* Wd1t = balloc(KW * DDIM * DDIM); bf16_t* Wd2t = balloc(KW * DDIM * DDIM); bf16_t* Wd3t = balloc(KW * DDIM * DDIM);
    if (off * sizeof(float) + boff * sizeof(bf16_t) > ws_size) return;

    float* out_a = (float*)d_out;
    float* out_v = out_a + NP * DA;
    float* out_d = out_v + NP * DV * 3;

    const int  EB = 256;
    auto EG = [&](size_t n) { return dim3((unsigned)((n + EB - 1) / EB)); };
    const dim3 TB(128);
    auto G  = [](int M, int Nc) { return dim3(M / 64, Nc / 32); };
    auto PW = [&](const float* W, bf16_t* Wt, int Kc, int Nc, int nt) {
        prep_wt_kernel<<<EG((size_t)nt * Kc * Nc), EB, 0, stream>>>(W, Wt, Kc, Nc, nt);
    };

    // ---- weight prep (transpose + bf16) ----
    PW(Waa, Waa_t, DA, DA, 1);     PW(Wda, Wda_t, DDIM, DA, 1);
    PW(Wvv, Wvv_t, DV, DV, 1);     PW(Wdd, Wdd_t, DDIM, DDIM, 1);
    PW(Wdt, Wdt_t, DDIM, DDIM, 1); PW(Wad, Wad_t, DA, DDIM, 1);
    PW(A1, A1t, DA, RR, 1);        PW(A2, A2t, DA, RR, 1);
    PW(V1, V1t, DV, RR, 1);        PW(V2, V2t, DV, RR, 1);
    PW(D1, D1t, DDIM, RR, 1);      PW(D2, D2t, DDIM, RR, 1);
    PW(Oaaa, Oaaa_t, RR, DA, 1);   PW(Oavv, Oavv_t, RR, DA, 1);   PW(Oadd, Oadd_t, RR, DA, 1);
    PW(Ovav, Ovav_t, RR, DV, 1);   PW(Ovdv, Ovdv_t, RR, DV, 1);
    PW(Odad, Odad_t, RR, DDIM, 1); PW(Odvv, Odvv_t, RR, DDIM, 1); PW(Oddd, Oddd_t, RR, DDIM, 1);
    PW(Wa1, Wa1t, DA, DA, KW);     PW(Wa2, Wa2t, DA, DA, KW);     PW(Wa3, Wa3t, DA, DA, KW);
    PW(Wv1, Wv1t, DV, DV, KW);     PW(Wv2, Wv2t, DV, DV, KW);     PW(Wv3, Wv3t, DV, DV, KW);
    PW(Wd1, Wd1t, DDIM, DDIM, KW); PW(Wd2, Wd2t, DDIM, DDIM, KW); PW(Wd3, Wd3t, DDIM, DDIM, KW);

    // ---- pack + trace ----
    pack_v_kernel<<<EG(NP * DV), EB, 0, stream>>>(x_v, vP);
    pack_d_kernel<<<EG(NP * DDIM), EB, 0, stream>>>(x_d, dP, dPT, tr);

    // ---- AVDFullLinearMix ----
    gemm_wmma<0><<<G(NPOS, DA), TB, 0, stream>>>(x_a, Waa_t, nullptr, aL, DA, DA);
    gemm_wmma<GF_ACC><<<G(NPOS, DA), TB, 0, stream>>>(tr, Wda_t, nullptr, aL, DDIM, DA);
    gemm_wmma<0><<<G(3 * NPOS, DV), TB, 0, stream>>>(vP, Wvv_t, nullptr, vL, DV, DV);
    gemm_wmma<0><<<G(9 * NPOS, DDIM), TB, 0, stream>>>(dP, Wdd_t, nullptr, dL, DDIM, DDIM);
    gemm_wmma<GF_ACC><<<G(9 * NPOS, DDIM), TB, 0, stream>>>(dPT, Wdt_t, nullptr, dL, DDIM, DDIM);
    gemm_wmma<0><<<G(NPOS, DDIM), TB, 0, stream>>>(x_a, Wad_t, nullptr, tad, DA, DDIM);
    add_diag_kernel<<<EG(NP * DDIM), EB, 0, stream>>>(dL, tad);

    // ---- low-rank projections ----
    gemm_wmma<0><<<G(NPOS, RR), TB, 0, stream>>>(aL, A1t, nullptr, pa1, DA, RR);
    gemm_wmma<0><<<G(NPOS, RR), TB, 0, stream>>>(aL, A2t, nullptr, pa2, DA, RR);
    gemm_wmma<0><<<G(3 * NPOS, RR), TB, 0, stream>>>(vL, V1t, nullptr, pv1, DV, RR);
    gemm_wmma<0><<<G(3 * NPOS, RR), TB, 0, stream>>>(vL, V2t, nullptr, pv2, DV, RR);
    gemm_wmma<0><<<G(9 * NPOS, RR), TB, 0, stream>>>(dL, D1t, nullptr, pd1, DDIM, RR);
    gemm_wmma<0><<<G(9 * NPOS, RR), TB, 0, stream>>>(dL, D2t, nullptr, pd2, DDIM, RR);

    // ---- bilinear contractions ----
    bilin_a_kernel<<<EG(NP * RR), EB, 0, stream>>>(pa1, pa2, pv1, pv2, pd1, pd2, sa, sv, sd);
    bilin_v_kernel<<<EG(NP * RR), EB, 0, stream>>>(pa1, pv2, pd1, tva, tvd);
    bilin_d_kernel<<<EG(NP * RR), EB, 0, stream>>>(pa1, pv1, pv2, pd1, pd2, tda, tdv, tdd);

    // ---- low-rank output maps + inner residual ----
    gemm_wmma<GF_CIN><<<G(NPOS, DA), TB, 0, stream>>>(sa, Oaaa_t, aL, a1, RR, DA);
    gemm_wmma<GF_ACC><<<G(NPOS, DA), TB, 0, stream>>>(sv, Oavv_t, nullptr, a1, RR, DA);
    gemm_wmma<GF_ACC><<<G(NPOS, DA), TB, 0, stream>>>(sd, Oadd_t, nullptr, a1, RR, DA);
    gemm_wmma<GF_CIN><<<G(3 * NPOS, DV), TB, 0, stream>>>(tva, Ovav_t, vL, v1, RR, DV);
    gemm_wmma<GF_ACC><<<G(3 * NPOS, DV), TB, 0, stream>>>(tvd, Ovdv_t, nullptr, v1, RR, DV);
    gemm_wmma<GF_CIN><<<G(9 * NPOS, DDIM), TB, 0, stream>>>(tda, Odad_t, dL, d1, RR, DDIM);
    gemm_wmma<GF_ACC><<<G(9 * NPOS, DDIM), TB, 0, stream>>>(tdv, Odvv_t, nullptr, d1, RR, DDIM);
    gemm_wmma<GF_ACC><<<G(9 * NPOS, DDIM), TB, 0, stream>>>(tdd, Oddd_t, nullptr, d1, RR, DDIM);

    // ---- a-stream conv MLP + GN + outer residual ----
    const dim3 cgA(BB * (LL / 64), DA / 32);
    conv_wmma<CF_BIAS | CF_LRELU><<<cgA, TB, 0, stream>>>(a1, Wa1t, ba1, nullptr, t1, DA);
    conv_wmma<CF_BIAS | CF_LRELU><<<cgA, TB, 0, stream>>>(t1, Wa2t, ba2, nullptr, t2, DA);
    conv_wmma<CF_BIAS | CF_RES><<<cgA, TB, 0, stream>>>(t2, Wa3t, ba3, a1, t1, DA);
    gn_a_kernel<<<EG(NP * 8), EB, 0, stream>>>(t1, x_a, ga, bA, out_a);

    // ---- v-stream conv MLP (tensor-sigmoid gates) + GN + outer residual ----
    const dim3 cgV(3 * BB * (LL / 64), DV / 32);
    conv_wmma<0><<<cgV, TB, 0, stream>>>(v1, Wv1t, nullptr, nullptr, t1, DV);
    tsig_kernel<<<EG(NP * DV), EB, 0, stream>>>(t1, DV, 3);
    conv_wmma<0><<<cgV, TB, 0, stream>>>(t1, Wv2t, nullptr, nullptr, t2, DV);
    tsig_kernel<<<EG(NP * DV), EB, 0, stream>>>(t2, DV, 3);
    conv_wmma<CF_RES><<<cgV, TB, 0, stream>>>(t2, Wv3t, nullptr, v1, t1, DV);
    gn_v_kernel<<<EG(NP * 8), EB, 0, stream>>>(t1, x_v, gv, out_v);

    // ---- d-stream conv MLP + GN + outer residual ----
    const dim3 cgD(9 * BB * (LL / 64), DDIM / 32);
    conv_wmma<0><<<cgD, TB, 0, stream>>>(d1, Wd1t, nullptr, nullptr, t1, DDIM);
    tsig_kernel<<<EG(NP * DDIM), EB, 0, stream>>>(t1, DDIM, 9);
    conv_wmma<0><<<cgD, TB, 0, stream>>>(t1, Wd2t, nullptr, nullptr, t2, DDIM);
    tsig_kernel<<<EG(NP * DDIM), EB, 0, stream>>>(t2, DDIM, 9);
    conv_wmma<CF_RES><<<cgD, TB, 0, stream>>>(t2, Wd3t, nullptr, d1, t1, DDIM);
    gn_d_kernel<<<EG(NP * 8), EB, 0, stream>>>(t1, x_d, gd, out_d);
}
